// CoordinateDecoder_11536282157071
// MI455X (gfx1250) — matrix operand order repaired and checked by reading, not compiled, not run on hardware
//
#include <hip/hip_runtime.h>
#include <hip/hip_bf16.h>

typedef __attribute__((ext_vector_type(16))) _Float16 v16h;
typedef __attribute__((ext_vector_type(8)))  _Float16 v8h;
typedef __attribute__((ext_vector_type(8)))  float    v8f;

#define BB     8
#define HH     64
#define WWG    64
#define CCH    64
#define NPTS   16384
#define MLPW   256
#define ENCD   42
#define INDIM  234
#define ROWS   64
#define STRIDE 264   // 256 + 8 halfs; 528B row stride -> conflict-free A-frag loads

// ---------------- prep kernels ----------------

__global__ void ctx_kernel(const float* __restrict__ cv, const float* __restrict__ Wc,
                           const float* __restrict__ bc, float* __restrict__ ctx) {
    int b = blockIdx.x;            // 8
    int j = threadIdx.x;           // 512
    float s = bc[j];
    #pragma unroll 8
    for (int c = 0; c < CCH; ++c) s += cv[b * CCH + c] * Wc[c * (2 * MLPW) + j];
    ctx[b * (2 * MLPW) + j] = s;
}

// 2x average-pool downsample (approximates jax.image.resize bilinear 2x down)
__global__ void down2_kernel(const float* __restrict__ src, float* __restrict__ dst,
                             int Hd, int Wd) {
    int i = blockIdx.x * blockDim.x + threadIdx.x;
    int total = BB * Hd * Wd * CCH;
    if (i >= total) return;
    int c = i & (CCH - 1);
    int t = i / CCH;
    int x = t % Wd; t /= Wd;
    int y = t % Hd;
    int b = t / Hd;
    int Ws = Wd * 2;
    const float* s = src + (((size_t)b * (Hd * 2) + 2 * y) * Ws + 2 * x) * CCH + c;
    dst[i] = 0.25f * (s[0] + s[CCH] + s[(size_t)Ws * CCH] + s[(size_t)Ws * CCH + CCH]);
}

// Wt[n][k] = (k<Kin && n<Nin) ? (f16)W[k][n] : 0   (transpose + pad + convert)
__global__ void wtT_kernel(const float* __restrict__ W, _Float16* __restrict__ Wt,
                           int Kin, int Nin, int Kpad, int Npad) {
    int i = blockIdx.x * blockDim.x + threadIdx.x;
    if (i >= Kpad * Npad) return;
    int k = i % Kpad;
    int n = i / Kpad;
    float v = (k < Kin && n < Nin) ? W[(size_t)k * Nin + n] : 0.0f;
    Wt[i] = (_Float16)v;
}

// ---------------- fused MLP kernel ----------------

// GELU via x*sigmoid(1.702x): native v_exp_f32 + v_rcp_f32, ~5 ops
__device__ __forceinline__ float gelu_fast(float x) {
    float e = __expf(-1.702f * x);
    return x * __builtin_amdgcn_rcpf(1.0f + e);
}

// tanh via 1 - 2/(e^{2x}+1): saturates correctly to +/-1
__device__ __forceinline__ float tanh_fast(float x) {
    float e = __expf(2.0f * x);
    return 1.0f - 2.0f * __builtin_amdgcn_rcpf(e + 1.0f);
}

__global__ __launch_bounds__(256)
void mlp_fused(const float* __restrict__ grid0, const float* __restrict__ pyr1,
               const float* __restrict__ pyr2,  const float* __restrict__ coords,
               const float* __restrict__ ctx,
               const _Float16* __restrict__ Wt0, const _Float16* __restrict__ Wt1,
               const _Float16* __restrict__ Wt2, const _Float16* __restrict__ Wt3,
               const _Float16* __restrict__ WtO,
               const float* __restrict__ b0, const float* __restrict__ b1,
               const float* __restrict__ b2, const float* __restrict__ b3,
               const float* __restrict__ bout,
               float* __restrict__ out) {
    __shared__ _Float16 act[ROWS * STRIDE];

    const int b   = blockIdx.y;
    const int p0  = blockIdx.x * ROWS;
    const int tid = threadIdx.x;

    // ---------- phase A: build 64x256 f16 input tile ----------
    {
        int r = tid >> 2;          // row 0..63
        int q = tid & 3;           // channel quarter
        int p = p0 + r;
        float cy = coords[(size_t)p * 2 + 0];
        float cx = coords[(size_t)p * 2 + 1];
        _Float16* row = &act[r * STRIDE];

        if (q == 0) {
            row[0] = (_Float16)cy;
            row[1] = (_Float16)cx;
            float fy = cy * 3.14159265358979f;
            float fx = cx * 3.14159265358979f;
            #pragma unroll
            for (int i = 0; i < 10; ++i) {
                row[2 + 4 * i + 0] = (_Float16)__sinf(fy);
                row[2 + 4 * i + 1] = (_Float16)__sinf(fx);
                row[2 + 4 * i + 2] = (_Float16)__cosf(fy);
                row[2 + 4 * i + 3] = (_Float16)__cosf(fx);
                fy *= 2.0f; fx *= 2.0f;
            }
        }
        if (q == 3) {
            #pragma unroll
            for (int k = INDIM; k < 256; ++k) row[k] = (_Float16)0.0f;
        }

        const float* levels[3] = {
            grid0 + (size_t)b * HH * WWG * CCH,
            pyr1  + (size_t)b * 32 * 32 * CCH,
            pyr2  + (size_t)b * 16 * 16 * CCH
        };
        const int cbase = q * 16;
        #pragma unroll
        for (int l = 0; l < 3; ++l) {
            int Hl = HH >> l, Wl = WWG >> l;
            float y = (cy + 1.0f) * 0.5f * (float)(Hl - 1);
            float x = (cx + 1.0f) * 0.5f * (float)(Wl - 1);
            float y0f = floorf(y), x0f = floorf(x);
            float wy = y - y0f, wx = x - x0f;
            int y0 = (int)y0f; y0 = y0 < 0 ? 0 : (y0 > Hl - 1 ? Hl - 1 : y0);
            int x0 = (int)x0f; x0 = x0 < 0 ? 0 : (x0 > Wl - 1 ? Wl - 1 : x0);
            int y1 = y0 + 1 > Hl - 1 ? Hl - 1 : y0 + 1;
            int x1 = x0 + 1 > Wl - 1 ? Wl - 1 : x0 + 1;
            float w00 = (1.0f - wx) * (1.0f - wy), w01 = wx * (1.0f - wy);
            float w10 = (1.0f - wx) * wy,          w11 = wx * wy;
            const float* g = levels[l];
            const float4* t00 = (const float4*)&g[((size_t)y0 * Wl + x0) * CCH + cbase];
            const float4* t01 = (const float4*)&g[((size_t)y0 * Wl + x1) * CCH + cbase];
            const float4* t10 = (const float4*)&g[((size_t)y1 * Wl + x0) * CCH + cbase];
            const float4* t11 = (const float4*)&g[((size_t)y1 * Wl + x1) * CCH + cbase];
            #pragma unroll
            for (int cc = 0; cc < 4; ++cc) {
                float4 a = t00[cc], bq = t01[cc], cq = t10[cc], dq = t11[cc];
                int k = ENCD + l * CCH + cbase + cc * 4;
                row[k + 0] = (_Float16)(a.x * w00 + bq.x * w01 + cq.x * w10 + dq.x * w11);
                row[k + 1] = (_Float16)(a.y * w00 + bq.y * w01 + cq.y * w10 + dq.y * w11);
                row[k + 2] = (_Float16)(a.z * w00 + bq.z * w01 + cq.z * w10 + dq.z * w11);
                row[k + 3] = (_Float16)(a.w * w00 + bq.w * w01 + cq.w * w10 + dq.w * w11);
            }
        }
    }
    __syncthreads();

    // ---------- phase B: 4 FiLM-MLP layers via WMMA ----------
    const int lane   = tid & 31;
    const int wid    = tid >> 5;
    const int mtile  = wid >> 1;            // 0..3 -> 16-row slab
    const int nhalf  = wid & 1;             // 0..1 -> 128-col half
    const int m0     = mtile * 16;
    const int arow   = m0 + (lane & 15);    // A: lane -> row M
    const int klane8 = (lane >> 4) * 8;     // A interleave: [kb..kb+7, kb+16..kb+23]
    const int nloc   = lane & 15;           // B/C: lane -> column N
    const int krowB  = (lane >> 4) * 8;     // B interleave: [kb..kb+7, kb+16..kb+23]
    const int mrow0  = m0 + (lane >> 4) * 8;

    const float* gam = &ctx[(size_t)b * (2 * MLPW)];
    const float* bet = gam + MLPW;

    #pragma unroll
    for (int layer = 0; layer < 4; ++layer) {
        const _Float16* Wl = (layer == 0) ? Wt0 : (layer == 1) ? Wt1 : (layer == 2) ? Wt2 : Wt3;
        const _Float16* Wn = (layer == 0) ? Wt1 : (layer == 1) ? Wt2 : (layer == 2) ? Wt3 : 0;
        const float*    bl = (layer == 0) ? b0  : (layer == 1) ? b1  : (layer == 2) ? b2  : b3;

        // A fragments: 8 k-tiles of 16x32 from LDS
        v16h afr[8];
        #pragma unroll
        for (int k = 0; k < 8; ++k) {
            const _Float16* base = &act[arow * STRIDE + k * 32 + klane8];
            ((v8h*)&afr[k])[0] = *(const v8h*)(base);
            ((v8h*)&afr[k])[1] = *(const v8h*)(base + 16);
        }

        // prefetch next layer's weight columns toward the WGP (L2 -> near caches)
        if (layer < 3) {
            #pragma unroll
            for (int nt = 0; nt < 8; ++nt) {
                int ng = nhalf * 128 + nt * 16 + nloc;
                __builtin_prefetch(&Wn[(size_t)ng * 256], 0, 0);
            }
        }

        v8f accs[8];
        #pragma unroll
        for (int nt = 0; nt < 8; ++nt) {
            int ng = nhalf * 128 + nt * 16 + nloc;
            v8f acc = {};
            #pragma unroll
            for (int k = 0; k < 8; ++k) {
                v16h bf;
                const _Float16* wb = &Wl[(size_t)ng * 256 + k * 32 + krowB];
                ((v8h*)&bf)[0] = *(const v8h*)(wb);
                ((v8h*)&bf)[1] = *(const v8h*)(wb + 16);
                acc = __builtin_amdgcn_wmma_f32_16x16x32_f16(
                        false, afr[k], false, bf, (short)0, acc, false, false);
            }
            accs[nt] = acc;
        }
        __syncthreads();   // all reads of act for this layer done

        #pragma unroll
        for (int nt = 0; nt < 8; ++nt) {
            int ng = nhalf * 128 + nt * 16 + nloc;
            float bias = bl[ng];
            float ga   = gam[ng] + 1.0f;
            float be   = bet[ng];
            #pragma unroll
            for (int j = 0; j < 8; ++j) {
                float x = accs[nt][j] + bias;
                x = x * ga + be;
                act[(mrow0 + j) * STRIDE + ng] = (_Float16)gelu_fast(x);
            }
        }
        __syncthreads();   // writes visible before next layer's reads
    }

    // ---------- phase C: output layer (256 -> 3, padded to 16) ----------
    if (nhalf == 0) {
        v16h afr[8];
        #pragma unroll
        for (int k = 0; k < 8; ++k) {
            const _Float16* base = &act[arow * STRIDE + k * 32 + klane8];
            ((v8h*)&afr[k])[0] = *(const v8h*)(base);
            ((v8h*)&afr[k])[1] = *(const v8h*)(base + 16);
        }
        v8f acc = {};
        #pragma unroll
        for (int k = 0; k < 8; ++k) {
            v16h bf;
            const _Float16* wb = &WtO[(size_t)nloc * 256 + k * 32 + krowB];
            ((v8h*)&bf)[0] = *(const v8h*)(wb);
            ((v8h*)&bf)[1] = *(const v8h*)(wb + 16);
            acc = __builtin_amdgcn_wmma_f32_16x16x32_f16(
                    false, afr[k], false, bf, (short)0, acc, false, false);
        }
        if (nloc < 3) {
            float bo = bout[nloc];
            #pragma unroll
            for (int j = 0; j < 8; ++j) {
                int p = p0 + mrow0 + j;
                out[((size_t)b * NPTS + p) * 3 + nloc] = tanh_fast(acc[j] + bo);
            }
        }
    }
}

// ---------------- launcher ----------------

extern "C" void kernel_launch(void* const* d_in, const int* in_sizes, int n_in,
                              void* d_out, int out_size, void* d_ws, size_t ws_size,
                              hipStream_t stream) {
    const float* feature_grid = (const float*)d_in[0];
    const float* context_vec  = (const float*)d_in[1];
    const float* coords       = (const float*)d_in[2];
    const float* Wc   = (const float*)d_in[3];
    const float* bc   = (const float*)d_in[4];
    const float* W0   = (const float*)d_in[5];
    const float* b0   = (const float*)d_in[6];
    const float* W1   = (const float*)d_in[7];
    const float* b1   = (const float*)d_in[8];
    const float* W2   = (const float*)d_in[9];
    const float* b2   = (const float*)d_in[10];
    const float* W3   = (const float*)d_in[11];
    const float* b3   = (const float*)d_in[12];
    const float* Wout = (const float*)d_in[13];
    const float* bout = (const float*)d_in[14];
    float* out = (float*)d_out;

    // workspace carve-up (all offsets 256B-aligned)
    char* ws = (char*)d_ws;
    size_t off = 0;
    float* ctx  = (float*)(ws + off); off += (size_t)BB * 2 * MLPW * 4;            // 16 KB
    float* pyr1 = (float*)(ws + off); off += (size_t)BB * 32 * 32 * CCH * 4;       // 2 MB
    float* pyr2 = (float*)(ws + off); off += (size_t)BB * 16 * 16 * CCH * 4;       // 512 KB
    _Float16* Wt0 = (_Float16*)(ws + off); off += (size_t)256 * 256 * 2;           // 128 KB
    _Float16* Wt1 = (_Float16*)(ws + off); off += (size_t)256 * 256 * 2;
    _Float16* Wt2 = (_Float16*)(ws + off); off += (size_t)256 * 256 * 2;
    _Float16* Wt3 = (_Float16*)(ws + off); off += (size_t)256 * 256 * 2;
    _Float16* WtO = (_Float16*)(ws + off); off += (size_t)16 * 256 * 2;            // 8 KB

    // context FiLM parameters
    ctx_kernel<<<dim3(BB), dim3(2 * MLPW), 0, stream>>>(context_vec, Wc, bc, ctx);

    // feature pyramid
    {
        int t1 = BB * 32 * 32 * CCH;
        down2_kernel<<<dim3((t1 + 255) / 256), dim3(256), 0, stream>>>(feature_grid, pyr1, 32, 32);
        int t2 = BB * 16 * 16 * CCH;
        down2_kernel<<<dim3((t2 + 255) / 256), dim3(256), 0, stream>>>(pyr1, pyr2, 16, 16);
    }

    // transposed f16 weights
    wtT_kernel<<<dim3(256 * 256 / 256), dim3(256), 0, stream>>>(W0,   Wt0, INDIM, 256, 256, 256);
    wtT_kernel<<<dim3(256 * 256 / 256), dim3(256), 0, stream>>>(W1,   Wt1, 256,   256, 256, 256);
    wtT_kernel<<<dim3(256 * 256 / 256), dim3(256), 0, stream>>>(W2,   Wt2, 256,   256, 256, 256);
    wtT_kernel<<<dim3(256 * 256 / 256), dim3(256), 0, stream>>>(W3,   Wt3, 256,   256, 256, 256);
    wtT_kernel<<<dim3(16 * 256 / 256),  dim3(256), 0, stream>>>(Wout, WtO, 256,   3,   256, 16);

    // fused MLP: 256 row-tiles per batch, 8 batches, 8 waves/WG
    mlp_fused<<<dim3(NPTS / ROWS, BB), dim3(256), 0, stream>>>(
        feature_grid, pyr1, pyr2, coords, ctx,
        Wt0, Wt1, Wt2, Wt3, WtO,
        b0, b1, b2, b3, bout, out);
}
